// FusionModel1D2D_12876311954184
// MI455X (gfx1250) — compile-verified, matrix operand order checked by reading, not run on hardware
//
#include <hip/hip_runtime.h>
#include <stdint.h>

typedef __attribute__((ext_vector_type(16))) __bf16 v16bf;
typedef __attribute__((ext_vector_type(8)))  float  v8f;

#define DEV_INLINE __device__ __forceinline__

// ---- fixed model dims (from reference) ----
#define NN 10000   // graph nodes
#define EE 100000  // edges (before self loops)
#define BB 64      // batch
#define LL 1000    // seq len
#define DDIM 150   // embedding dim

// ---------------- helpers ----------------
DEV_INLINE __bf16 f2bf(float f) {
  unsigned u = __builtin_bit_cast(unsigned, f);
  unsigned r = (u + 0x7FFFu + ((u >> 16) & 1u)) >> 16;
  unsigned short s = (unsigned short)r;
  return __builtin_bit_cast(__bf16, s);
}
DEV_INLINE unsigned orderedU(float f) {
  unsigned u = __builtin_bit_cast(unsigned, f);
  return (u & 0x80000000u) ? ~u : (u | 0x80000000u);
}
DEV_INLINE float unorderedF(unsigned u) {
  unsigned v = (u & 0x80000000u) ? (u & 0x7FFFFFFFu) : ~u;
  return __builtin_bit_cast(float, v);
}
DEV_INLINE void edge_sd(const int* ei, int eid, int& s, int& d) {
  if (eid < EE) { s = ei[eid]; d = ei[EE + eid]; }
  else          { s = d = eid - EE; }   // self loops appended
}

// ---------------- bf16 conversions ----------------
__global__ void to_bf16(const float* __restrict__ src, __bf16* __restrict__ dst, int n) {
  int i = blockIdx.x * blockDim.x + threadIdx.x;
  if (i < n) dst[i] = f2bf(src[i]);
}
// transpose-convert: src [K,N] row-major fp32 -> dst [N,K] row-major bf16
__global__ void to_bf16_t(const float* __restrict__ src, __bf16* __restrict__ dst,
                          int K, int N) {
  int idx = blockIdx.x * blockDim.x + threadIdx.x;
  if (idx >= K * N) return;
  int n = idx / K;
  int k = idx - n * K;
  dst[idx] = f2bf(src[(size_t)k * N + n]);
}

// ---- fragment load: 16 bf16 (K-block of 32) for one row, per ISA 7.12.2 wave32 layout
// lane<16: VGPR p holds K=kb+2p,kb+2p+1 (kb=0); VGPR 4+p holds K=16+kb+2p. lanes>=16: kb=8.
DEV_INLINE v16bf load_frag(const __bf16* __restrict__ row, int k0, int kb) {
  v16bf v;
#pragma unroll
  for (int p = 0; p < 4; ++p) {
    int ka  = k0 + kb + 2 * p;
    int ka2 = ka + 16;
    v[2*p]     = row[ka];
    v[2*p + 1] = row[ka + 1];
    v[2*p + 8] = row[ka2];
    v[2*p + 9] = row[ka2 + 1];
  }
  return v;
}

// ---------------- WMMA bf16 GEMM ----------------
// C[M,N] = A[M,K] @ B[K,N]  where Bt is B transposed: Bt[N,K] row-major bf16.
// M%16==0, N%64==0, K%32==0 guaranteed by caller.
// One wave computes a 16x64 tile: shared A fragment feeds 4 WMMA accumulators.
__global__ __launch_bounds__(256)
void gemm_bf16_wmma(const __bf16* __restrict__ A, const __bf16* __restrict__ Bt,
                    const float* __restrict__ bias, float* __restrict__ C,
                    int M, int N, int K, int relu) {
  const int lane   = threadIdx.x & 31;
  const int wave   = threadIdx.x >> 5;
  const int tilesN = N >> 6;                       // 64-wide wave tiles
  const int tilesM = M >> 4;
  int tile = blockIdx.x * (blockDim.x >> 5) + wave;
  if (tile >= tilesM * tilesN) return;             // uniform per wave: EXEC stays all-ones
  const int tM = tile / tilesN;
  const int tN = tile - tM * tilesN;
  const int mrow  = (tM << 4) + (lane & 15);
  const int nbase = tN << 6;
  const int kb    = (lane >> 4) << 3;              // 0 (lanes 0-15) / 8 (lanes 16-31)
  const __bf16* Arow = A + (size_t)mrow * K;
  const __bf16* Br0  = Bt + (size_t)(nbase +      (lane & 15)) * K;
  const __bf16* Br1  = Br0 + (size_t)16 * K;
  const __bf16* Br2  = Br0 + (size_t)32 * K;
  const __bf16* Br3  = Br0 + (size_t)48 * K;
  v8f acc0 = {}, acc1 = {}, acc2 = {}, acc3 = {};
  for (int k0 = 0; k0 < K; k0 += 32) {
    __builtin_prefetch(Arow + k0 + 128, 0, 1);     // global_prefetch_b8 (speculative)
    __builtin_prefetch(Br0  + k0 + 128, 0, 1);
    v16bf a  = load_frag(Arow, k0, kb);
    v16bf b0 = load_frag(Br0,  k0, kb);
    v16bf b1 = load_frag(Br1,  k0, kb);
    v16bf b2 = load_frag(Br2,  k0, kb);
    v16bf b3 = load_frag(Br3,  k0, kb);
    acc0 = __builtin_amdgcn_wmma_f32_16x16x32_bf16(false, a, false, b0, (short)0, acc0, false, false);
    acc1 = __builtin_amdgcn_wmma_f32_16x16x32_bf16(false, a, false, b1, (short)0, acc1, false, false);
    acc2 = __builtin_amdgcn_wmma_f32_16x16x32_bf16(false, a, false, b2, (short)0, acc2, false, false);
    acc3 = __builtin_amdgcn_wmma_f32_16x16x32_bf16(false, a, false, b3, (short)0, acc3, false, false);
  }
  // C layout: VGPR r -> M = r + 8*(lane>=16), N = (lane&15) within each 16-col tile
  v8f accs[4] = {acc0, acc1, acc2, acc3};
#pragma unroll
  for (int j = 0; j < 4; ++j) {
    int ncol = nbase + 16 * j + (lane & 15);
    float bv = bias ? bias[ncol] : 0.0f;
#pragma unroll
    for (int r = 0; r < 8; ++r) {
      int row = (tM << 4) + r + ((lane >> 4) << 3);
      float v = accs[j][r] + bv;
      if (relu) v = fmaxf(v, 0.0f);
      C[(size_t)row * N + ncol] = v;
    }
  }
}

// ---------------- CNN branch ----------------
__global__ void embed_kernel(const int* __restrict__ x1d, const float* __restrict__ emb,
                             float* __restrict__ t0) {
  int idx = blockIdx.x * blockDim.x + threadIdx.x;
  if (idx >= BB * DDIM * LL) return;
  int l = idx % LL;
  int d = (idx / LL) % DDIM;
  int b = idx / (DDIM * LL);
  int tok = x1d[b * LL + l];
  t0[idx] = emb[tok * DDIM + d];
}

__global__ void conv1_kernel(const float* __restrict__ t0, const float* __restrict__ w,
                             const float* __restrict__ bias, float* __restrict__ out) {
  int idx = blockIdx.x * blockDim.x + threadIdx.x;
  if (idx >= BB * 64 * LL) return;
  int l  = idx % LL;
  int co = (idx / LL) & 63;
  int b  = idx / (64 * LL);
  const float* xb = t0 + (size_t)b * DDIM * LL;
  const float* wc = w + co * DDIM * 3;
  float acc = bias[co];
  for (int ci = 0; ci < DDIM; ++ci) {
    const float* xr = xb + ci * LL;
    float xm = (l > 0)      ? xr[l - 1] : 0.0f;
    float xc = xr[l];
    float xp = (l < LL - 1) ? xr[l + 1] : 0.0f;
    acc = fmaf(wc[3 * ci],     xm, acc);
    acc = fmaf(wc[3 * ci + 1], xc, acc);
    acc = fmaf(wc[3 * ci + 2], xp, acc);
  }
  out[idx] = acc;
}

__global__ void bnc_stats(const float* __restrict__ x, float* __restrict__ mean,
                          float* __restrict__ var) {
  int c = blockIdx.x;
  float s = 0.f, s2 = 0.f;
  for (int i = threadIdx.x; i < BB * LL; i += blockDim.x) {
    int b = i / LL, l = i - (i / LL) * LL;
    float v = x[(size_t)b * 64 * LL + c * LL + l];
    s += v; s2 += v * v;
  }
  __shared__ float sh[256], sh2[256];
  sh[threadIdx.x] = s; sh2[threadIdx.x] = s2;
  __syncthreads();
  for (int st = 128; st > 0; st >>= 1) {
    if ((int)threadIdx.x < st) {
      sh[threadIdx.x]  += sh[threadIdx.x + st];
      sh2[threadIdx.x] += sh2[threadIdx.x + st];
    }
    __syncthreads();
  }
  if (threadIdx.x == 0) {
    float m = sh[0] / (float)(BB * LL);
    mean[c] = m;
    var[c]  = sh2[0] / (float)(BB * LL) - m * m;
  }
}

__global__ void pool_bn_relu(const float* __restrict__ c1, const float* __restrict__ mean,
                             const float* __restrict__ var, const float* __restrict__ g,
                             const float* __restrict__ be, float* __restrict__ p1) {
  int idx = blockIdx.x * blockDim.x + threadIdx.x;
  if (idx >= BB * 64 * 500) return;
  int l2 = idx % 500;
  int c  = (idx / 500) & 63;
  int b  = idx / (64 * 500);
  float m = mean[c];
  float inv = rsqrtf(var[c] + 1e-5f);
  float gg = g[c], bb = be[c];
  size_t base = (size_t)b * 64 * LL + c * LL + 2 * l2;
  float a0 = gg * (c1[base]     - m) * inv + bb;
  float a1 = gg * (c1[base + 1] - m) * inv + bb;
  p1[idx] = fmaxf(fmaxf(a0, a1), 0.0f);
}

__global__ void conv2_kernel(const float* __restrict__ p1, const float* __restrict__ w,
                             const float* __restrict__ bias, float* __restrict__ out) {
  int idx = blockIdx.x * blockDim.x + threadIdx.x;
  if (idx >= BB * 128 * 500) return;
  int l  = idx % 500;
  int co = (idx / 500) & 127;
  int b  = idx / (128 * 500);
  const float* xb = p1 + (size_t)b * 64 * 500;
  const float* wc = w + co * 64 * 3;
  float acc = bias[co];
  for (int ci = 0; ci < 64; ++ci) {
    const float* xr = xb + ci * 500;
    float xm = (l > 0)   ? xr[l - 1] : 0.0f;
    float xc = xr[l];
    float xp = (l < 499) ? xr[l + 1] : 0.0f;
    acc = fmaf(wc[3 * ci],     xm, acc);
    acc = fmaf(wc[3 * ci + 1], xc, acc);
    acc = fmaf(wc[3 * ci + 2], xp, acc);
  }
  out[idx] = fmaxf(acc, 0.0f);
}

// ---------------- GAT branch ----------------
__global__ void gat_scores(const float* __restrict__ h, const float* __restrict__ as,
                           const float* __restrict__ ad, float* __restrict__ ss,
                           float* __restrict__ sd, int H) {
  int idx = blockIdx.x * blockDim.x + threadIdx.x;
  if (idx >= NN * H) return;
  int n  = idx / H;
  int hh = idx - n * H;
  const float* hr  = h + (size_t)n * H * 128 + hh * 128;
  const float* asr = as + hh * 128;
  const float* adr = ad + hh * 128;
  float a = 0.f, d = 0.f;
  for (int c = 0; c < 128; ++c) {
    float v = hr[c];
    a = fmaf(v, asr[c], a);
    d = fmaf(v, adr[c], d);
  }
  ss[idx] = a;
  sd[idx] = d;
}

__global__ void gat_edge1(const int* __restrict__ ei, const float* __restrict__ ss,
                          const float* __restrict__ sd, float* __restrict__ ee,
                          unsigned* __restrict__ mb, int H) {
  int idx = blockIdx.x * blockDim.x + threadIdx.x;
  if (idx >= (EE + NN) * H) return;
  int eid = idx / H, h = idx - (idx / H) * H;
  int s, d; edge_sd(ei, eid, s, d);
  float e = ss[s * H + h] + sd[d * H + h];
  e = (e > 0.0f) ? e : 0.2f * e;
  ee[idx] = e;
  atomicMax(&mb[d * H + h], orderedU(e));
}

__global__ void gat_edge2(const int* __restrict__ ei, float* __restrict__ ee,
                          const unsigned* __restrict__ mb, float* __restrict__ den, int H) {
  int idx = blockIdx.x * blockDim.x + threadIdx.x;
  if (idx >= (EE + NN) * H) return;
  int eid = idx / H, h = idx - (idx / H) * H;
  int s, d; edge_sd(ei, eid, s, d);
  float m = unorderedF(mb[d * H + h]);
  float w = __expf(ee[idx] - m);
  ee[idx] = w;
  atomicAdd(&den[d * H + h], w);
}

__global__ void gat_alpha(const int* __restrict__ ei, float* __restrict__ ee,
                          const float* __restrict__ den, int H) {
  int idx = blockIdx.x * blockDim.x + threadIdx.x;
  if (idx >= (EE + NN) * H) return;
  int eid = idx / H, h = idx - (idx / H) * H;
  int s, d; edge_sd(ei, eid, s, d);
  ee[idx] = ee[idx] / (den[d * H + h] + 1e-16f);
}

// weighted scatter: hout[dst] += hmid[src] * alpha ; float4 per thread (C=128 -> f>>7 = head)
__global__ void gat_edge3(const int* __restrict__ ei, const float* __restrict__ alpha,
                          const float* __restrict__ hmid, float* __restrict__ hout,
                          int H, int HC) {
  int idx = blockIdx.x * blockDim.x + threadIdx.x;
  int q = HC >> 2;
  int total = (EE + NN) * q;
  if (idx >= total) return;
  int eid = idx / q;
  int f4  = (idx - eid * q) << 2;
  int s, d; edge_sd(ei, eid, s, d);
  float a = alpha[eid * H + (f4 >> 7)];
  const float4 hv = *(const float4*)(hmid + (size_t)s * HC + f4);
  float* o = hout + (size_t)d * HC + f4;
  atomicAdd(o + 0, hv.x * a);
  atomicAdd(o + 1, hv.y * a);
  atomicAdd(o + 2, hv.z * a);
  atomicAdd(o + 3, hv.w * a);
}

__global__ void col_stats(const float* __restrict__ x, float* __restrict__ mean,
                          float* __restrict__ var, int Nrows, int F) {
  int f = blockIdx.x;
  float s = 0.f, s2 = 0.f;
  for (int i = threadIdx.x; i < Nrows; i += blockDim.x) {
    float v = x[(size_t)i * F + f];
    s += v; s2 += v * v;
  }
  __shared__ float sh[256], sh2[256];
  sh[threadIdx.x] = s; sh2[threadIdx.x] = s2;
  __syncthreads();
  for (int st = 128; st > 0; st >>= 1) {
    if ((int)threadIdx.x < st) {
      sh[threadIdx.x]  += sh[threadIdx.x + st];
      sh2[threadIdx.x] += sh2[threadIdx.x + st];
    }
    __syncthreads();
  }
  if (threadIdx.x == 0) {
    float m = sh[0] / (float)Nrows;
    mean[f] = m;
    var[f]  = sh2[0] / (float)Nrows - m * m;
  }
}

__global__ void bn_relu(float* __restrict__ x, const float* __restrict__ mean,
                        const float* __restrict__ var, const float* __restrict__ g,
                        const float* __restrict__ b, int total, int F) {
  int idx = blockIdx.x * blockDim.x + threadIdx.x;
  if (idx >= total) return;
  int f = idx % F;
  float v = g[f] * (x[idx] - mean[f]) * rsqrtf(var[f] + 1e-5f) + b[f];
  x[idx] = fmaxf(v, 0.0f);
}

// ---------------- pooling + tail ----------------
__global__ void pool_cnt(const int* __restrict__ batch, float* __restrict__ cnt) {
  int n = blockIdx.x * blockDim.x + threadIdx.x;
  if (n < NN) atomicAdd(&cnt[batch[n]], 1.0f);
}
__global__ void pool_sum(const float* __restrict__ h, const int* __restrict__ batch,
                         float* __restrict__ pool) {
  int idx = blockIdx.x * blockDim.x + threadIdx.x;
  if (idx >= NN * 128) return;
  int n = idx >> 7, f = idx & 127;
  atomicAdd(&pool[batch[n] * 128 + f], h[idx]);
}
__global__ void pool_div(const float* __restrict__ pool, const float* __restrict__ cnt,
                         float* __restrict__ hg) {
  int idx = blockIdx.x * blockDim.x + threadIdx.x;
  if (idx >= BB * 128) return;
  int b = idx >> 7;
  hg[idx] = pool[idx] / fmaxf(cnt[b], 1.0f);
}

__global__ void fc_relu(const float* __restrict__ A, const float* __restrict__ W,
                        const float* __restrict__ bias, float* __restrict__ out,
                        int M, int K, int Nn, int relu) {
  int idx = blockIdx.x * blockDim.x + threadIdx.x;
  if (idx >= M * Nn) return;
  int m = idx / Nn, n = idx - m * Nn;
  float acc = bias ? bias[n] : 0.0f;
  for (int k = 0; k < K; ++k) acc = fmaf(A[m * K + k], W[k * Nn + n], acc);
  if (relu) acc = fmaxf(acc, 0.0f);
  out[idx] = acc;
}

__global__ void fusion_out(const float* __restrict__ x1, const float* __restrict__ hg,
                           const float* __restrict__ a1p, const float* __restrict__ a2p,
                           const float* __restrict__ fcw, const float* __restrict__ fcb,
                           float* __restrict__ out) {
  int idx = blockIdx.x * blockDim.x + threadIdx.x;
  if (idx >= BB * 4) return;
  int m = idx / 4, j = idx - m * 4;
  float a1 = a1p[0], a2 = a2p[0];
  float acc = fcb[j];
  for (int f = 0; f < 256; ++f)
    acc = fmaf(a1 * x1[m * 256 + f] + a2 * hg[m * 256 + f], fcw[f * 4 + j], acc);
  out[idx] = acc;
}

// ---------------- workspace arena (float units) ----------------
static const size_t OFF_T0   = 0;                       // 9,600,000
static const size_t OFF_C1   = OFF_T0 + 9600000;        // 4,096,000
static const size_t OFF_P1   = OFF_C1 + 4096000;        // 2,048,000
static const size_t OFF_C2   = OFF_P1 + 2048000;        // 4,096,000
static const size_t OFF_C2BF = OFF_C2 + 4096000;        // 2,048,000 (bf16 slots)
static const size_t OFF_WFC  = OFF_C2BF + 2048000;      // 8,192,000 (bf16 slots)
static const size_t CNN_END  = OFF_WFC + 8192000;       // 30,080,000
// GAT phase aliases the CNN region (CNN is finished by then):
static const size_t OFF_H0   = 0;                       // 5,120,000
static const size_t OFF_H1   = OFF_H0 + 5120000;
static const size_t OFF_HMID = OFF_H1 + 5120000;
static const size_t OFF_XBF  = OFF_HMID + 5120000;      // 2,560,000 (bf16 slots)
static const size_t OFF_WBF  = OFF_XBF + 2560000;       // 131,072  (bf16 slots)
static const size_t OFF_EEB  = OFF_WBF + 131072;        // 440,000
static const size_t OFF_SS   = OFF_EEB + 440000;        // 40,000
static const size_t OFF_SD   = OFF_SS + 40000;
static const size_t OFF_MB   = OFF_SD + 40000;
static const size_t OFF_DEN  = OFF_MB + 40000;
// persistent (past both phases):
static const size_t OFF_X1   = CNN_END;                 // 16,384
static const size_t OFF_MEAN = OFF_X1 + 16384;          // 512
static const size_t OFF_VAR  = OFF_MEAN + 512;          // 512
static const size_t OFF_POOL = OFF_VAR + 512;           // 8,192
static const size_t OFF_CNT  = OFF_POOL + 8192;         // 64
static const size_t OFF_HG   = OFF_CNT + 64;            // 8,192
static const size_t OFF_F1O  = OFF_HG + 8192;           // 8,192
static const size_t OFF_F2O  = OFF_F1O + 8192;          // 16,384

static unsigned nblk(long long n) { return (unsigned)((n + 255) / 256); }

extern "C" void kernel_launch(void* const* d_in, const int* in_sizes, int n_in,
                              void* d_out, int out_size, void* d_ws, size_t ws_size,
                              hipStream_t stream) {
  (void)in_sizes; (void)n_in; (void)out_size; (void)ws_size;
  float* ws = (float*)d_ws;

  const int*   x1d   = (const int*)d_in[0];
  const float* xnode = (const float*)d_in[1];
  const int*   eidx  = (const int*)d_in[2];
  const int*   batch = (const int*)d_in[3];
  const float* emb   = (const float*)d_in[4];
  const float* c1w   = (const float*)d_in[5];
  const float* c1b   = (const float*)d_in[6];
  const float* bn1dg = (const float*)d_in[7];
  const float* bn1db = (const float*)d_in[8];
  const float* c2w   = (const float*)d_in[9];
  const float* c2b   = (const float*)d_in[10];
  const float* fc1dw = (const float*)d_in[11];
  const float* fc1db = (const float*)d_in[12];
  const float* g1w   = (const float*)d_in[13];
  const float* g1as  = (const float*)d_in[14];
  const float* g1ad  = (const float*)d_in[15];
  const float* bn1g  = (const float*)d_in[17];
  const float* bn1b  = (const float*)d_in[18];
  const float* g26w  = (const float*)d_in[19];
  const float* g26as = (const float*)d_in[20];
  const float* g26ad = (const float*)d_in[21];
  const float* bn26g = (const float*)d_in[23];
  const float* bn26b = (const float*)d_in[24];
  const float* g7w   = (const float*)d_in[25];
  const float* g7as  = (const float*)d_in[26];
  const float* g7ad  = (const float*)d_in[27];
  const float* bn7g  = (const float*)d_in[29];
  const float* bn7b  = (const float*)d_in[30];
  const float* fc1w  = (const float*)d_in[31];
  const float* fc1b  = (const float*)d_in[32];
  const float* fc2dw = (const float*)d_in[33];
  const float* fc2db = (const float*)d_in[34];
  const float* att1  = (const float*)d_in[35];
  const float* att2  = (const float*)d_in[36];
  const float* fcw   = (const float*)d_in[37];
  const float* fcb   = (const float*)d_in[38];

  const int TB = 256;

  // ===== CNN branch =====
  embed_kernel<<<nblk((long long)BB * DDIM * LL), TB, 0, stream>>>(x1d, emb, ws + OFF_T0);
  conv1_kernel<<<nblk((long long)BB * 64 * LL), TB, 0, stream>>>(ws + OFF_T0, c1w, c1b, ws + OFF_C1);
  bnc_stats<<<64, TB, 0, stream>>>(ws + OFF_C1, ws + OFF_MEAN, ws + OFF_VAR);
  pool_bn_relu<<<nblk((long long)BB * 64 * 500), TB, 0, stream>>>(
      ws + OFF_C1, ws + OFF_MEAN, ws + OFF_VAR, bn1dg, bn1db, ws + OFF_P1);
  conv2_kernel<<<nblk((long long)BB * 128 * 500), TB, 0, stream>>>(ws + OFF_P1, c2w, c2b, ws + OFF_C2);
  to_bf16<<<nblk((long long)BB * 64000), TB, 0, stream>>>(ws + OFF_C2, (__bf16*)(ws + OFF_C2BF), BB * 64000);
  to_bf16_t<<<nblk(64000LL * 256), TB, 0, stream>>>(fc1dw, (__bf16*)(ws + OFF_WFC), 64000, 256);
  {
    int tiles = (BB / 16) * (256 / 64);
    gemm_bf16_wmma<<<(tiles + 7) / 8, TB, 0, stream>>>(
        (__bf16*)(ws + OFF_C2BF), (__bf16*)(ws + OFF_WFC), fc1db, ws + OFF_X1,
        BB, 256, 64000, 1);
  }

  // ===== GAT branch =====
  float* hbuf[2] = { ws + OFF_H0, ws + OFF_H1 };
  const float* Warr[7]; const float* asArr[7]; const float* adArr[7];
  const float* bgArr[7]; const float* bbArr[7];
  int cinArr[7] = {128, 512, 512, 512, 512, 512, 512};
  int Harr[7]   = {4, 4, 4, 4, 4, 4, 1};
  Warr[0] = g1w; asArr[0] = g1as; adArr[0] = g1ad; bgArr[0] = bn1g; bbArr[0] = bn1b;
  for (int i = 0; i < 5; ++i) {
    Warr[1 + i]  = g26w + (size_t)i * 512 * 512;
    asArr[1 + i] = g26as + i * 4 * 128;
    adArr[1 + i] = g26ad + i * 4 * 128;
    bgArr[1 + i] = bn26g + i * 512;
    bbArr[1 + i] = bn26b + i * 512;
  }
  Warr[6] = g7w; asArr[6] = g7as; adArr[6] = g7ad; bgArr[6] = bn7g; bbArr[6] = bn7b;

  const float* hin = xnode;
  const int E2 = EE + NN;
  for (int layer = 0; layer < 7; ++layer) {
    int Cin = cinArr[layer], H = Harr[layer], HC = H * 128;
    __bf16* xbf = (__bf16*)(ws + OFF_XBF);
    __bf16* wbf = (__bf16*)(ws + OFF_WBF);
    float* hmid = ws + OFF_HMID;
    float* hout = hbuf[layer & 1];

    to_bf16<<<nblk((long long)NN * Cin), TB, 0, stream>>>(hin, xbf, NN * Cin);
    to_bf16_t<<<nblk((long long)Cin * HC), TB, 0, stream>>>(Warr[layer], wbf, Cin, HC);
    {
      int tiles = (NN / 16) * (HC / 64);
      gemm_bf16_wmma<<<(tiles + 7) / 8, TB, 0, stream>>>(xbf, wbf, nullptr, hmid,
                                                         NN, HC, Cin, 0);
    }
    gat_scores<<<nblk((long long)NN * H), TB, 0, stream>>>(
        hmid, asArr[layer], adArr[layer], ws + OFF_SS, ws + OFF_SD, H);
    hipMemsetAsync(ws + OFF_MB,  0, (size_t)NN * H * sizeof(float), stream);
    hipMemsetAsync(ws + OFF_DEN, 0, (size_t)NN * H * sizeof(float), stream);
    gat_edge1<<<nblk((long long)E2 * H), TB, 0, stream>>>(
        eidx, ws + OFF_SS, ws + OFF_SD, ws + OFF_EEB, (unsigned*)(ws + OFF_MB), H);
    gat_edge2<<<nblk((long long)E2 * H), TB, 0, stream>>>(
        eidx, ws + OFF_EEB, (const unsigned*)(ws + OFF_MB), ws + OFF_DEN, H);
    gat_alpha<<<nblk((long long)E2 * H), TB, 0, stream>>>(eidx, ws + OFF_EEB, ws + OFF_DEN, H);
    hipMemsetAsync(hout, 0, (size_t)NN * HC * sizeof(float), stream);
    gat_edge3<<<nblk((long long)E2 * (HC >> 2)), TB, 0, stream>>>(
        eidx, ws + OFF_EEB, hmid, hout, H, HC);
    col_stats<<<HC, TB, 0, stream>>>(hout, ws + OFF_MEAN, ws + OFF_VAR, NN, HC);
    bn_relu<<<nblk((long long)NN * HC), TB, 0, stream>>>(
        hout, ws + OFF_MEAN, ws + OFF_VAR, bgArr[layer], bbArr[layer], NN * HC, HC);
    hin = hout;
  }

  // ===== global mean pool + tail FCs =====
  hipMemsetAsync(ws + OFF_POOL, 0, BB * 128 * sizeof(float), stream);
  hipMemsetAsync(ws + OFF_CNT,  0, BB * sizeof(float), stream);
  pool_cnt<<<nblk(NN), TB, 0, stream>>>(batch, ws + OFF_CNT);
  pool_sum<<<nblk((long long)NN * 128), TB, 0, stream>>>(hin, batch, ws + OFF_POOL);
  pool_div<<<nblk(BB * 128), TB, 0, stream>>>(ws + OFF_POOL, ws + OFF_CNT, ws + OFF_HG);
  fc_relu<<<nblk(BB * 128), TB, 0, stream>>>(ws + OFF_HG, fc1w, fc1b, ws + OFF_F1O,
                                             BB, 128, 128, 1);
  fc_relu<<<nblk(BB * 256), TB, 0, stream>>>(ws + OFF_F1O, fc2dw, fc2db, ws + OFF_F2O,
                                             BB, 128, 256, 1);
  fusion_out<<<nblk(BB * 4), TB, 0, stream>>>(ws + OFF_X1, ws + OFF_F2O, att1, att2,
                                              fcw, fcb, (float*)d_out);
}
